// OmniGenMLP_29515015258325
// MI455X (gfx1250) — compile-verified, matrix operand order checked by reading, not run on hardware
//
#include <hip/hip_runtime.h>

// ---------------------------------------------------------------------------
// SwiGLU MLP for MI455X (gfx1250, wave32).
//   hidden = silu(x @ Wg + bg) * (x @ Wu + bu)      [4096 x 8192]  (bf16, d_ws)
//   out    = hidden @ Wd + bd                        [4096 x 2048]  (fp32)
// Compute-bound (AI ~1.5 kFLOP/B) -> V_WMMA_F32_16X16X32_BF16, fp32 accum.
// Kernel 2 A-tiles staged with GLOBAL_LOAD_ASYNC_TO_LDS_B128 (ASYNCcnt).
// ---------------------------------------------------------------------------

typedef __attribute__((ext_vector_type(16))) __bf16 v16bf;
typedef __attribute__((ext_vector_type(8)))  float  v8f;

#define M_TOT 4096
#define D_DIM 2048
#define I_DIM 8192

// LDS tile stride: 64 K-elements + 8 pad (bf16) = 72 -> 36 uints, multiple of 4
// so every fragment read is a 16B-aligned ds_load_b128.
#define KSTRIDE_U 36

__device__ __forceinline__ unsigned short f2bf(float f) {
  unsigned u = __float_as_uint(f);
  u += 0x7FFFu + ((u >> 16) & 1u);   // round-to-nearest-even
  return (unsigned short)(u >> 16);
}

#if __has_builtin(__builtin_amdgcn_cvt_pk_bf16_f32)
typedef __attribute__((ext_vector_type(2))) __bf16 v2bf;
__device__ __forceinline__ unsigned pack2(float lo, float hi) {
  union { v2bf v; unsigned u; } c;
  c.v = __builtin_amdgcn_cvt_pk_bf16_f32(lo, hi);
  return c.u;
}
#else
__device__ __forceinline__ unsigned pack2(float lo, float hi) {
  return (unsigned)f2bf(lo) | ((unsigned)f2bf(hi) << 16);
}
#endif

// Load one 16x32 bf16 A/B fragment from an LDS tile stored [row][k] with
// KSTRIDE_U uint stride. Per ISA 16-bit layout: lane -> row = lane%16,
// half = lane/16; VGPR v holds K = kk + (v&3)*2 + half*8 + (v>=4)*16.
__device__ __forceinline__ v16bf ld_frag(const unsigned* lds, int row, int kk, int half) {
  union { v16bf v; uint4 q[2]; } u;
  const uint4* p = (const uint4*)(lds + row * KSTRIDE_U + (kk >> 1) + half * 4);
  u.q[0] = p[0];   // VGPRs 0..3  (K = kk+half*8   .. +7)
  u.q[1] = p[2];   // VGPRs 4..7  (K = kk+half*8+16 .. +23)
  return u.v;
}

// ---------------------------------------------------------------------------
// Kernel 1: fused gate/up GEMM + SwiGLU, bf16 hidden output.
// Block tile: M=128, N_hidden=64 (64 gate cols + 64 up cols).
// 8 waves: wm in {0,1} (64 rows), wn in {0..3} (16 hidden cols each).
// ---------------------------------------------------------------------------
__global__ __launch_bounds__(256)
void k_gateup(const float* __restrict__ x, const float* __restrict__ wgu,
              const float* __restrict__ bgu, unsigned short* __restrict__ hid) {
  __shared__ unsigned Au[128 * KSTRIDE_U];   // A tile  [m][k]  128x64 bf16
  __shared__ unsigned Bg[64  * KSTRIDE_U];   // gate B  [n][k]   64x64 bf16 (transposed)
  __shared__ unsigned Bu[64  * KSTRIDE_U];   // up   B  [n][k]

  const int tid  = threadIdx.x;
  const int lane = tid & 31;
  const int wave = tid >> 5;
  const int wm   = wave >> 2;      // 0..1
  const int wn   = wave & 3;       // 0..3
  const int half = lane >> 4;
  const int nl   = lane & 15;

  const int n0 = blockIdx.x * 64;    // hidden-column tile
  const int m0 = blockIdx.y * 128;   // token tile

  const v8f vzero = {0.f,0.f,0.f,0.f,0.f,0.f,0.f,0.f};
  v8f accg[4], accu[4];
#pragma unroll
  for (int i = 0; i < 4; ++i) { accg[i] = vzero; accu[i] = vzero; }

  // staging roles (coalesced: consecutive lanes -> consecutive addresses)
  const int srow  = tid >> 1;         // 0..127 (A row)
  const int shalf = tid & 1;          // which 32-wide K half
  const int bk    = tid >> 2;         // 0..63: B K-row (4 threads share a row)
  const int bn    = (tid & 3) * 16;   // 16-col group within the row

  for (int k0 = 0; k0 < D_DIM; k0 += 64) {
    // ---- stage A: x[m0+srow][k0 + shalf*32 .. +31] fp32 -> bf16 ----
    {
      const float4* src = (const float4*)(x + (size_t)(m0 + srow) * D_DIM + k0 + shalf * 32);
      uint4* dst = (uint4*)(Au + srow * KSTRIDE_U + shalf * 16);
#pragma unroll
      for (int q = 0; q < 4; ++q) {
        float4 a = src[2 * q], b = src[2 * q + 1];
        uint4 o;
        o.x = pack2(a.x, a.y); o.y = pack2(a.z, a.w);
        o.z = pack2(b.x, b.y); o.w = pack2(b.z, b.w);
        dst[q] = o;
      }
    }
    // ---- stage B gate/up: wgu[k0+bk][n0+bn .. +15] -> transposed [n][k] ----
    {
      const float* gsrc = wgu + (size_t)(k0 + bk) * (2 * I_DIM) + n0 + bn;
      const float* usrc = gsrc + I_DIM;
      unsigned short* bg16 = (unsigned short*)Bg;
      unsigned short* bu16 = (unsigned short*)Bu;
#pragma unroll
      for (int c = 0; c < 16; c += 4) {
        float4 g = *(const float4*)(gsrc + c);
        float4 u = *(const float4*)(usrc + c);
        bg16[(bn + c + 0) * (2 * KSTRIDE_U) + bk] = f2bf(g.x);
        bg16[(bn + c + 1) * (2 * KSTRIDE_U) + bk] = f2bf(g.y);
        bg16[(bn + c + 2) * (2 * KSTRIDE_U) + bk] = f2bf(g.z);
        bg16[(bn + c + 3) * (2 * KSTRIDE_U) + bk] = f2bf(g.w);
        bu16[(bn + c + 0) * (2 * KSTRIDE_U) + bk] = f2bf(u.x);
        bu16[(bn + c + 1) * (2 * KSTRIDE_U) + bk] = f2bf(u.y);
        bu16[(bn + c + 2) * (2 * KSTRIDE_U) + bk] = f2bf(u.z);
        bu16[(bn + c + 3) * (2 * KSTRIDE_U) + bk] = f2bf(u.w);
      }
    }
    // prefetch next K-tile (fire-and-forget -> global_prefetch_b8)
    if (k0 + 64 < D_DIM) {
      __builtin_prefetch(x + (size_t)(m0 + srow) * D_DIM + (k0 + 64) + shalf * 32, 0, 0);
      __builtin_prefetch(wgu + (size_t)(k0 + 64 + bk) * (2 * I_DIM) + n0 + bn, 0, 0);
      __builtin_prefetch(wgu + (size_t)(k0 + 64 + bk) * (2 * I_DIM) + I_DIM + n0 + bn, 0, 0);
    }
    __syncthreads();

#pragma unroll
    for (int kk = 0; kk < 64; kk += 32) {
      v16bf a[4];
#pragma unroll
      for (int i = 0; i < 4; ++i)
        a[i] = ld_frag(Au, wm * 64 + i * 16 + nl, kk, half);
      v16bf bg = ld_frag(Bg, wn * 16 + nl, kk, half);
      v16bf bu = ld_frag(Bu, wn * 16 + nl, kk, half);
#pragma unroll
      for (int i = 0; i < 4; ++i) {
        accg[i] = __builtin_amdgcn_wmma_f32_16x16x32_bf16(false, a[i], false, bg,
                                                          (short)0, accg[i], false, false);
        accu[i] = __builtin_amdgcn_wmma_f32_16x16x32_bf16(false, a[i], false, bu,
                                                          (short)0, accu[i], false, false);
      }
    }
    __syncthreads();
  }

  // ---- epilogue: bias + SwiGLU, store bf16 hidden ----
  const int nh = n0 + wn * 16 + nl;          // hidden column (D layout: N = lane%16)
  const float biasg = bgu[nh];
  const float biasu = bgu[I_DIM + nh];
#pragma unroll
  for (int i = 0; i < 4; ++i) {
#pragma unroll
    for (int r = 0; r < 8; ++r) {
      const int m = m0 + wm * 64 + i * 16 + r + half * 8;   // D layout: M = r + half*8
      const float g = accg[i][r] + biasg;
      const float u = accu[i][r] + biasu;
      const float h = g * u / (1.f + __expf(-g));           // silu(g) * u
      hid[(size_t)m * I_DIM + nh] = f2bf(h);
    }
  }
}

// ---------------------------------------------------------------------------
// Kernel 2: out = hidden(bf16) @ w_down + b_down.  Block tile 128x128,
// 8 waves 2x4: wave tile M=64 (4 frags), N=32 (2 frags).
// A-tiles staged via async global->LDS DMA (no VGPR round trip).
// ---------------------------------------------------------------------------
__global__ __launch_bounds__(256)
void k_down(const unsigned short* __restrict__ hid, const float* __restrict__ wdn,
            const float* __restrict__ bdn, float* __restrict__ out) {
  __shared__ unsigned Au[128 * KSTRIDE_U];   // A tile [m][k]  128x64 bf16
  __shared__ unsigned Bt[128 * KSTRIDE_U];   // B tile [n][k]  128x64 bf16 (transposed)

  const int tid  = threadIdx.x;
  const int lane = tid & 31;
  const int wave = tid >> 5;
  const int wm   = wave >> 2;
  const int wn   = wave & 3;
  const int half = lane >> 4;
  const int nl   = lane & 15;

  const int n0 = blockIdx.x * 128;
  const int m0 = blockIdx.y * 128;

  const v8f vzero = {0.f,0.f,0.f,0.f,0.f,0.f,0.f,0.f};
  v8f acc[4][2];
#pragma unroll
  for (int i = 0; i < 4; ++i) { acc[i][0] = vzero; acc[i][1] = vzero; }

  const int srow  = tid >> 1;
  const int shalf = tid & 1;
  const int bk    = tid >> 2;         // 0..63 (4 threads per row -> coalesced)
  const int bn    = (tid & 3) * 32;   // 32-col group

  // per-thread fixed LDS destination for the async A copy
  const unsigned ldsA =
      (unsigned)(uintptr_t)(Au + srow * KSTRIDE_U + shalf * 16);

  for (int k0 = 0; k0 < I_DIM; k0 += 64) {
    // ---- stage A: bf16 hidden -> LDS via async DMA (ASYNCcnt) ----
    {
      unsigned long long ga =
          (unsigned long long)(uintptr_t)(hid + (size_t)(m0 + srow) * I_DIM + k0 + shalf * 32);
#pragma unroll
      for (int q = 0; q < 4; ++q) {
        asm volatile("global_load_async_to_lds_b128 %0, %1, off"
                     :: "v"(ldsA + 16u * q), "v"(ga + 16ull * q)
                     : "memory");
      }
    }
    // ---- stage B: w_down[k0+bk][n0+bn .. +31] fp32 -> bf16, transposed ----
    {
      const float* bsrc = wdn + (size_t)(k0 + bk) * D_DIM + n0 + bn;
      unsigned short* bt16 = (unsigned short*)Bt;
#pragma unroll
      for (int c = 0; c < 32; c += 4) {
        float4 v = *(const float4*)(bsrc + c);
        bt16[(bn + c + 0) * (2 * KSTRIDE_U) + bk] = f2bf(v.x);
        bt16[(bn + c + 1) * (2 * KSTRIDE_U) + bk] = f2bf(v.y);
        bt16[(bn + c + 2) * (2 * KSTRIDE_U) + bk] = f2bf(v.z);
        bt16[(bn + c + 3) * (2 * KSTRIDE_U) + bk] = f2bf(v.w);
      }
    }
    if (k0 + 64 < I_DIM) {
      __builtin_prefetch(wdn + (size_t)(k0 + 64 + bk) * D_DIM + n0 + bn, 0, 0);
    }
    // wait for this wave's async LDS writes, then rendezvous
    asm volatile("s_wait_asynccnt 0x0" ::: "memory");
    __syncthreads();

#pragma unroll
    for (int kk = 0; kk < 64; kk += 32) {
      v16bf a[4], b[2];
#pragma unroll
      for (int i = 0; i < 4; ++i)
        a[i] = ld_frag(Au, wm * 64 + i * 16 + nl, kk, half);
#pragma unroll
      for (int j = 0; j < 2; ++j)
        b[j] = ld_frag(Bt, wn * 32 + j * 16 + nl, kk, half);
#pragma unroll
      for (int i = 0; i < 4; ++i)
#pragma unroll
        for (int j = 0; j < 2; ++j)
          acc[i][j] = __builtin_amdgcn_wmma_f32_16x16x32_bf16(false, a[i], false, b[j],
                                                              (short)0, acc[i][j], false, false);
    }
    __syncthreads();
  }

  // ---- epilogue: bias add, fp32 store ----
#pragma unroll
  for (int j = 0; j < 2; ++j) {
    const int n = n0 + wn * 32 + j * 16 + nl;
    const float bias = bdn[n];
#pragma unroll
    for (int i = 0; i < 4; ++i) {
#pragma unroll
      for (int r = 0; r < 8; ++r) {
        const int m = m0 + wm * 64 + i * 16 + r + half * 8;
        out[(size_t)m * D_DIM + n] = acc[i][j][r] + bias;
      }
    }
  }
}

// ---------------------------------------------------------------------------
extern "C" void kernel_launch(void* const* d_in, const int* in_sizes, int n_in,
                              void* d_out, int out_size, void* d_ws, size_t ws_size,
                              hipStream_t stream) {
  const float* x   = (const float*)d_in[0];   // [4096, 2048]
  const float* wgu = (const float*)d_in[1];   // [2048, 16384]
  const float* bgu = (const float*)d_in[2];   // [16384]
  const float* wdn = (const float*)d_in[3];   // [8192, 2048]
  const float* bdn = (const float*)d_in[4];   // [2048]
  float* out = (float*)d_out;                 // [4096, 2048]
  unsigned short* hid = (unsigned short*)d_ws; // [4096, 8192] bf16 = 64 MB

  dim3 g1(I_DIM / 64, M_TOT / 128);   // 128 x 32
  k_gateup<<<g1, 256, 0, stream>>>(x, wgu, bgu, hid);

  dim3 g2(D_DIM / 128, M_TOT / 128);  // 16 x 32
  k_down<<<g2, 256, 0, stream>>>(hid, wdn, bdn, out);
}
